// S_LSTMLayer_22273700397276
// MI455X (gfx1250) — compile-verified, matrix operand order checked by reading, not run on hardware
//
#include <hip/hip_runtime.h>
#include <hip/hip_bf16.h>

#define BATCH 64
#define SEQLEN 256
#define XS 512
#define OS 512      // OUT_SIZE
#define ND 2048     // 4*OUT_SIZE

typedef __attribute__((ext_vector_type(16))) __bf16       v16bf;
typedef __attribute__((ext_vector_type(8)))  float        v8f;
typedef __attribute__((ext_vector_type(4)))  unsigned int u32x4;
typedef __attribute__((ext_vector_type(4)))  float        f32x4;
typedef __attribute__((ext_vector_type(4)))  __bf16       bf16x4;

__device__ __forceinline__ __bf16 f2bf(float f) {
    unsigned u = __float_as_uint(f);
    unsigned r = (u + 0x7FFFu + ((u >> 16) & 1u)) >> 16;   // round-to-nearest-even
    unsigned short s = (unsigned short)r;
    __bf16 b;
    __builtin_memcpy(&b, &s, 2);
    return b;
}

__device__ __forceinline__ float sigm(float x)  { return 1.0f / (1.0f + __expf(-x)); }
__device__ __forceinline__ float tanh_(float x) { return 2.0f / (1.0f + __expf(-2.0f * x)) - 1.0f; }

union ABu { v16bf v; u32x4 q[2]; };

// Per-lane loader for WMMA bf16 A (16x32, row-major M x K source) and
// B (32x16, read from row-major N x K transposed weights). ISA layout:
// lanes 0-15 hold K in {k0+0..7, k0+16..23}; lanes 16-31 hold {k0+8..15, k0+24..31}.
__device__ __forceinline__ v16bf load_tile(const __bf16* __restrict__ base,
                                           int row0, int rowStride, int k0, int lane) {
    const int half = (lane >> 4) & 1;
    const int r = row0 + (lane & 15);
    const __bf16* p = base + (long)r * rowStride + k0 + half * 8;
    ABu u;
    u.q[0] = *(const u32x4*)(p);        // K = k0+c0 .. +7   (b128)
    u.q[1] = *(const u32x4*)(p + 16);   // K = k0+c0+16 .. +23 (b128)
    return u.v;
}

__device__ __forceinline__ v8f wmma_bf16(v16bf a, v16bf b, v8f c) {
    return __builtin_amdgcn_wmma_f32_16x16x32_bf16(false, a, false, b, (short)0, c, false, false);
}

// ---------------- one-time conversion kernels ----------------

// WT layout: 3 contiguous sections of [N=2048][K=512] bf16:
// sec0 = Wx^T (W rows 0..511), sec1 = Wh^T (rows 512..1023), sec2 = Wg^T (rows 1024..1535)
__global__ void convert_weights_kernel(const float* __restrict__ W, __bf16* __restrict__ WT) {
    long idx = (long)blockIdx.x * 256 + threadIdx.x;
    const long TOT = 3L * 2048 * 512;
    if (idx >= TOT) return;
    int  sec = (int)(idx / (2048L * 512));
    long rem = idx - (long)sec * 2048 * 512;
    int  n   = (int)(rem >> 9);
    int  k   = (int)(rem & 511);
    WT[idx] = f2bf(W[((long)sec * 512 + k) * ND + n]);
}

// x: fp32 (B, SEQ, X) -> Xb: bf16 TIME-MAJOR (SEQ, B, X) so each step's x_t
// slice is one contiguous 64 KB block. g: fp32 (B, 512) -> gb bf16 (identity).
__global__ void convert_inputs_kernel(const float* __restrict__ x, const float* __restrict__ g,
                                      __bf16* __restrict__ Xb, __bf16* __restrict__ gb) {
    long i = (long)blockIdx.x * 256 + threadIdx.x;          // group-of-4 elements
    const long NX4 = (long)BATCH * SEQLEN * XS / 4;
    const long NG4 = (long)BATCH * 512 / 4;
    if (i < NX4) {
        // output index: ((t*BATCH + b)*XS + k), k4 = k/4 contiguous
        long k4  = i & (XS / 4 - 1);
        long tb  = i >> 7;                                  // / (XS/4)
        long bI  = tb & (BATCH - 1);
        long tI  = tb >> 6;                                 // / BATCH
        long src = ((bI * SEQLEN + tI) * XS) / 4 + k4;      // fully coalesced in k
        f32x4 v = ((const f32x4*)x)[src];
        bf16x4 o; o[0]=f2bf(v[0]); o[1]=f2bf(v[1]); o[2]=f2bf(v[2]); o[3]=f2bf(v[3]);
        ((bf16x4*)Xb)[i] = o;
    } else if (i < NX4 + NG4) {
        long j = i - NX4;
        f32x4 v = ((const f32x4*)g)[j];
        bf16x4 o; o[0]=f2bf(v[0]); o[1]=f2bf(v[1]); o[2]=f2bf(v[2]); o[3]=f2bf(v[3]);
        ((bf16x4*)gb)[j] = o;
    }
}

__global__ void zero_kernel(unsigned int* __restrict__ p, long nwords) {
    long i = (long)blockIdx.x * 256 + threadIdx.x;
    if (i < nwords) p[i] = 0u;
}

// ---------------- g @ Wg + b  (time-invariant) ----------------
// 512 waves: wave -> (mtile 0..3, ntile 0..127); one 16x16 fp32 tile, K=512.
__global__ __launch_bounds__(256) void gproj_kernel(const __bf16* __restrict__ gb,
                                                    const __bf16* __restrict__ WgT,
                                                    const float*  __restrict__ bias,
                                                    float*        __restrict__ gWb) {
    const int tid = threadIdx.x;
    const int lane = tid & 31;
    const int w = blockIdx.x * 8 + (tid >> 5);          // 0..511
    const int m0 = (w >> 7) * 16;
    const int n0 = (w & 127) * 16;

    v8f acc = {};
#pragma unroll 4
    for (int kb = 0; kb < 16; ++kb) {
        const int k0 = kb * 32;
        v16bf a  = load_tile(gb,  m0, 512, k0, lane);
        v16bf bm = load_tile(WgT, n0, 512, k0, lane);
        acc = wmma_bf16(a, bm, acc);
    }
    const int nl = lane & 15;
    const int mh = (lane >> 4) * 8;
    const int n  = n0 + nl;
    const float bn = bias[n];
#pragma unroll
    for (int vg = 0; vg < 8; ++vg) {
        const int m = m0 + mh + vg;
        gWb[(long)m * ND + n] = acc[vg] + bn;
    }
}

// ---------------- recurrent step (both directions) ----------------
// 256 waves: dir (1b) | mtile (2b) | ntile (5b). Each wave: 16x16 patch of all
// 4 gates over K=1024 (x_t part + h part) = 128 v_wmma, then full LSTM update.
__global__ __launch_bounds__(256) void lstm_step_kernel(
        const __bf16* __restrict__ Xb,   const __bf16* __restrict__ WxT,
        const __bf16* __restrict__ WhT,  const float*  __restrict__ gWb,
        const __bf16* __restrict__ hIn,  __bf16*       __restrict__ hOut,
        float* __restrict__ cbuf,        float* __restrict__ out,
        float* __restrict__ hState,      float* __restrict__ cState, int s)
{
    const int tid   = threadIdx.x;
    const int lane  = tid & 31;
    const int w     = blockIdx.x * 8 + (tid >> 5);      // 0..255
    const int dir   = w >> 7;
    const int m0    = ((w >> 5) & 3) * 16;
    const int n0    = (w & 31) * 16;
    const int t     = dir ? (SEQLEN - 1 - s) : s;

    const __bf16* hDir  = hIn + dir * (BATCH * OS);
    const __bf16* xBase = Xb + (long)t * (BATCH * XS);  // time-major: row stride = XS

    __builtin_prefetch(hDir + (long)m0 * OS, 0, 0);
    __builtin_prefetch(xBase + (long)m0 * XS, 0, 0);

    v8f a0 = {}, a1 = {}, a2 = {}, a3 = {};

    // K = 0..511 : x_t @ Wx
#pragma unroll 4
    for (int kb = 0; kb < 16; ++kb) {
        const int k0 = kb * 32;
        v16bf av = load_tile(xBase, m0, XS, k0, lane);
        v16bf b0 = load_tile(WxT, n0,          512, k0, lane);
        v16bf b1 = load_tile(WxT, 512  + n0,   512, k0, lane);
        v16bf b2 = load_tile(WxT, 1024 + n0,   512, k0, lane);
        v16bf b3 = load_tile(WxT, 1536 + n0,   512, k0, lane);
        a0 = wmma_bf16(av, b0, a0);
        a1 = wmma_bf16(av, b1, a1);
        a2 = wmma_bf16(av, b2, a2);
        a3 = wmma_bf16(av, b3, a3);
    }
    // K = 512..1023 : h @ Wh
#pragma unroll 4
    for (int kb = 0; kb < 16; ++kb) {
        const int k0 = kb * 32;
        v16bf av = load_tile(hDir, m0, 512, k0, lane);
        v16bf b0 = load_tile(WhT, n0,          512, k0, lane);
        v16bf b1 = load_tile(WhT, 512  + n0,   512, k0, lane);
        v16bf b2 = load_tile(WhT, 1024 + n0,   512, k0, lane);
        v16bf b3 = load_tile(WhT, 1536 + n0,   512, k0, lane);
        a0 = wmma_bf16(av, b0, a0);
        a1 = wmma_bf16(av, b1, a1);
        a2 = wmma_bf16(av, b2, a2);
        a3 = wmma_bf16(av, b3, a3);
    }

    // elementwise LSTM update — owner-computes, fp32
    const int nl = lane & 15;
    const int mh = (lane >> 4) * 8;
    float*  cDir    = cbuf + dir * (BATCH * OS);
    __bf16* hOutDir = hOut + dir * (BATCH * OS);
    const bool last = (s == SEQLEN - 1);
#pragma unroll
    for (int vg = 0; vg < 8; ++vg) {
        const int m = m0 + mh + vg;
        const int n = n0 + nl;
        const float* gw = gWb + (long)m * ND;   // g@Wg + b, time-invariant
        const float zi = a0[vg] + gw[n];
        const float zf = a1[vg] + gw[512  + n];
        const float zo = a2[vg] + gw[1024 + n];
        const float zu = a3[vg] + gw[1536 + n];
        const long ci = (long)m * OS + n;
        const float cold = cDir[ci];
        const float cnew = sigm(zf) * cold + sigm(zi) * tanh_(zu);
        const float hnew = sigm(zo) * tanh_(cnew);
        cDir[ci] = cnew;
        hOutDir[ci] = f2bf(hnew);
        out[((long)m * SEQLEN + t) * (2 * OS) + dir * OS + n] = hnew;
        if (last) {
            const long si = (long)m * (2 * OS) + dir * OS + n;
            hState[si] = hnew;
            cState[si] = cnew;
        }
    }
}

// ---------------- host side ----------------
extern "C" void kernel_launch(void* const* d_in, const int* in_sizes, int n_in,
                              void* d_out, int out_size, void* d_ws, size_t ws_size,
                              hipStream_t stream) {
    const float* x = (const float*)d_in[0];
    const float* g = (const float*)d_in[1];
    const float* W = (const float*)d_in[2];
    const float* b = (const float*)d_in[3];

    float* out    = (float*)d_out;
    float* hState = out + (long)BATCH * SEQLEN * (2 * OS);
    float* cState = hState + (long)BATCH * (2 * OS);

    // workspace layout (all offsets 256B-aligned), total ~24 MiB
    char* ws = (char*)d_ws;
    const size_t szWT  = 3L * 2048 * 512 * 2;                 // 6 MiB (Wx^T|Wh^T|Wg^T bf16)
    const size_t szXb  = (long)BATCH * SEQLEN * XS * 2;       // 16 MiB (time-major)
    const size_t szGb  = (long)BATCH * 512 * 2;               // 64 KiB
    const size_t szGWb = (long)BATCH * ND * 4;                // 512 KiB
    const size_t szH   = 2L * 2 * BATCH * OS * 2;             // 256 KiB (2 parity x 2 dir)
    __bf16* WT   = (__bf16*)(ws);
    __bf16* WxT  = WT;
    __bf16* WhT  = WT + 2048L * 512;
    __bf16* WgT  = WT + 2L * 2048 * 512;
    __bf16* Xb   = (__bf16*)(ws + szWT);
    __bf16* gb   = (__bf16*)(ws + szWT + szXb);
    float*  gWb  = (float*) (ws + szWT + szXb + szGb);
    __bf16* hbuf = (__bf16*)(ws + szWT + szXb + szGb + szGWb);
    float*  cbuf = (float*) (ws + szWT + szXb + szGb + szGWb + szH);

    // 1) weights fp32 -> bf16, transposed to N-major
    {
        long tot = 3L * 2048 * 512;
        convert_weights_kernel<<<dim3((unsigned)((tot + 255) / 256)), dim3(256), 0, stream>>>(W, WT);
    }
    // 2) x (to time-major), g fp32 -> bf16
    {
        long tot4 = (long)BATCH * SEQLEN * XS / 4 + (long)BATCH * 512 / 4;
        convert_inputs_kernel<<<dim3((unsigned)((tot4 + 255) / 256)), dim3(256), 0, stream>>>(x, g, Xb, gb);
    }
    // 3) zero h (both parities) and c — contiguous 512 KiB = 131072 dwords
    zero_kernel<<<dim3(512), dim3(256), 0, stream>>>((unsigned int*)hbuf, 131072L);
    // 4) gWb = g @ Wg + b  (time-invariant)
    gproj_kernel<<<dim3(64), dim3(256), 0, stream>>>(gb, WgT, b, gWb);
    // 5) 256 sequential steps, both directions per launch; h double-buffered
    const long PH = 2L * BATCH * OS;   // parity slab stride (bf16 elements)
    for (int s = 0; s < SEQLEN; ++s) {
        const __bf16* hIn = hbuf + (s & 1) * PH;
        __bf16* hOut      = hbuf + ((s + 1) & 1) * PH;
        lstm_step_kernel<<<dim3(32), dim3(256), 0, stream>>>(
            Xb, WxT, WhT, gWb, hIn, hOut, cbuf, out, hState, cState, s);
    }
    (void)in_sizes; (void)n_in; (void)out_size; (void)ws_size;
}
